// SelfAttention_13486197309914
// MI455X (gfx1250) — matrix operand order repaired and checked by reading, not compiled, not run on hardware
//
#include <hip/hip_runtime.h>
#include <hip/hip_bf16.h>

// ---------------------------------------------------------------------------
// Self-attention (B=2,S=2048,DIM=1536,H=12,HD=128) for gfx1250 / MI455X.
// Matmuls: v_wmma_f32_16x16x32_f16.  Tile staging: global_load_async_to_lds
// (ASYNCcnt) with double-buffered LDS, single barrier per pipeline stage.
// ---------------------------------------------------------------------------

#define B_    2
#define S_    2048
#define DIM_  1536
#define H_    12
#define HD_   128
#define M_    (B_ * S_)          // 4096 tokens
#define NPAIR 64                 // HD/2 rope pairs per head

typedef _Float16 half8  __attribute__((ext_vector_type(8)));
typedef _Float16 half16 __attribute__((ext_vector_type(16)));
typedef float    f32x8  __attribute__((ext_vector_type(8)));

static __device__ __forceinline__ half16 cat16(half8 lo, half8 hi) {
    return __builtin_shufflevector(lo, hi, 0,1,2,3,4,5,6,7,8,9,10,11,12,13,14,15);
}

static __device__ __forceinline__ f32x8 wmma_f16(half16 a, half16 b, f32x8 c) {
    return __builtin_amdgcn_wmma_f32_16x16x32_f16(
        /*neg_a=*/false, a, /*neg_b=*/false, b,
        /*c_mod=*/(short)0, c, /*reuse_a=*/false, /*reuse_b=*/false);
}

// Async 16B global->LDS copy (per lane), tracked by ASYNCcnt.
// LDS aperture addresses truncate to the workgroup-relative LDS offset, so the
// low 32 bits of the generic pointer are the VDST value the instruction wants.
static __device__ __forceinline__ void async_ld_b128(void* lds_p, const void* gp) {
    uint32_t l = (uint32_t)(uintptr_t)lds_p;
    asm volatile("global_load_async_to_lds_b128 %0, %1, off"
                 :: "v"(l), "v"(gp) : "memory");
}
static __device__ __forceinline__ void wait_async0() {
    asm volatile("s_wait_asynccnt 0" ::: "memory");
}

// ---------------------------------------------------------------------------
// 0) f32 -> f16 conversion
// ---------------------------------------------------------------------------
__global__ void cvt_f32_to_f16(const float* __restrict__ src,
                               _Float16* __restrict__ dst, int n) {
    int i = blockIdx.x * blockDim.x + threadIdx.x;
    if (i < n) dst[i] = (_Float16)src[i];
}

// ---------------------------------------------------------------------------
// 1) NT GEMM: C[M,N] = A[M,K] * B[N,K]^T + bias[N]   (f16 in, f32 out)
//    BM=BN=128, BK=32, 256 threads = 8 waves; wave tile 64x32 (4x2 WMMA tiles)
//    Double-buffered LDS fed by async global->LDS copies; one barrier/k-step.
// ---------------------------------------------------------------------------
#define BM 128
#define BN 128
#define BK 32
#define KP 8   // LDS row pad (f16) -> row stride 40 f16 = 80 B (16B multiple)

__global__ __launch_bounds__(256)
void gemm_nt_f16_f32(const _Float16* __restrict__ A,   // [M,K] row-major
                     const _Float16* __restrict__ Bw,  // [N,K] row-major
                     const float*    __restrict__ bias,// [N]
                     float*          __restrict__ C,   // [M,N] row-major
                     int M, int N, int K)
{
    __shared__ __align__(16) _Float16 As[2][BM][BK + KP];
    __shared__ __align__(16) _Float16 Bs[2][BN][BK + KP];

    const int tid  = threadIdx.x;
    const int lane = tid & 31;
    const int wave = tid >> 5;
    const int m_blk = blockIdx.y * BM;
    const int n_blk = blockIdx.x * BN;
    const int wm = (wave >> 2) * 64;   // wave M offset: 0 / 64
    const int wn = (wave & 3) * 32;    // wave N offset: 0,32,64,96

    f32x8 acc[4][2];
#pragma unroll
    for (int mt = 0; mt < 4; ++mt)
#pragma unroll
        for (int nt = 0; nt < 2; ++nt)
#pragma unroll
            for (int r = 0; r < 8; ++r) acc[mt][nt][r] = 0.0f;

    const int lrow  = tid & 127;        // tile row this thread copies
    const int lhalf = (tid >> 7) * 16;  // k-offset 0 or 16 (16 f16 each)

    const int arow = (lane & 15);
    const int kh8  = (lane >> 4) * 8;   // A-frag k sub-offset
    const int kh16 = (lane >> 4) * 16;  // unused (B-frag also uses kh8)

    auto issue_tile = [&](int k0, int buf) {
        const _Float16* ag = A + (size_t)(m_blk + lrow) * K + k0 + lhalf;
        async_ld_b128(&As[buf][lrow][lhalf],     ag);
        async_ld_b128(&As[buf][lrow][lhalf + 8], ag + 8);
        const _Float16* bg = Bw + (size_t)(n_blk + lrow) * K + k0 + lhalf;
        async_ld_b128(&Bs[buf][lrow][lhalf],     bg);
        async_ld_b128(&Bs[buf][lrow][lhalf + 8], bg + 8);
    };

    issue_tile(0, 0);
    int cur = 0;
    for (int k0 = 0; k0 < K; k0 += BK) {
        wait_async0();          // my async copies for buffer `cur` done
        __syncthreads();        // everyone's copies visible; prev reads done
        if (k0 + BK < K) issue_tile(k0 + BK, cur ^ 1);

        half16 af[4];
#pragma unroll
        for (int mt = 0; mt < 4; ++mt) {
            const int r = wm + mt * 16 + arow;
            af[mt] = cat16(*(const half8*)&As[cur][r][kh8],
                           *(const half8*)&As[cur][r][16 + kh8]);
        }
        half16 bf[2];
#pragma unroll
        for (int nt = 0; nt < 2; ++nt) {
            const int r = wn + nt * 16 + arow;
            bf[nt] = cat16(*(const half8*)&Bs[cur][r][kh8],
                           *(const half8*)&Bs[cur][r][16 + kh8]);
        }
#pragma unroll
        for (int mt = 0; mt < 4; ++mt)
#pragma unroll
            for (int nt = 0; nt < 2; ++nt)
                acc[mt][nt] = wmma_f16(af[mt], bf[nt], acc[mt][nt]);
        cur ^= 1;
        // no trailing barrier: next iteration's top barrier orders
        // "all waves done reading buf" before "async refill of buf visible".
    }

    // --- epilogue: C layout (lane = n, VGPR r = row) + bias
#pragma unroll
    for (int nt = 0; nt < 2; ++nt) {
        const int n = n_blk + wn + nt * 16 + (lane & 15);
        const float bv = bias[n];
#pragma unroll
        for (int mt = 0; mt < 4; ++mt) {
#pragma unroll
            for (int r = 0; r < 8; ++r) {
                const int m = m_blk + wm + mt * 16 + r + (lane >> 4) * 8;
                C[(size_t)m * N + n] = acc[mt][nt][r] + bv;
            }
        }
    }
}

// ---------------------------------------------------------------------------
// 2) RMS-norm (full DIM) + RoPE on Q,K; layout shuffle + f16 cast for Q,K,V
//    qh,kh: [B,H,S,HD] f16 ; vth: [B,H,HD,S] f16 (transposed for PV WMMA)
// ---------------------------------------------------------------------------
__global__ __launch_bounds__(256)
void rmsrope_qkv(const float* __restrict__ qf, const float* __restrict__ kf,
                 const float* __restrict__ vf, const float* __restrict__ freqs,
                 const float* __restrict__ gq, const float* __restrict__ gk,
                 _Float16* __restrict__ qh, _Float16* __restrict__ kh,
                 _Float16* __restrict__ vth)
{
    const int tok = blockIdx.x;          // 0..B*S-1
    const int b = tok / S_;
    const int s = tok % S_;
    const int tid = threadIdx.x;
    const int lane = tid & 31;
    const int wave = tid >> 5;

    const float* qrow = qf + (size_t)tok * DIM_;
    const float* krow = kf + (size_t)tok * DIM_;

    float sq = 0.f, sk = 0.f;
    for (int i = tid; i < DIM_; i += 256) {
        float a = qrow[i]; sq += a * a;
        float c = krow[i]; sk += c * c;
    }
#pragma unroll
    for (int m = 16; m >= 1; m >>= 1) {
        sq += __shfl_xor(sq, m);
        sk += __shfl_xor(sk, m);
    }
    __shared__ float redq[8], redk[8];
    if (lane == 0) { redq[wave] = sq; redk[wave] = sk; }
    __syncthreads();
    float tq = 0.f, tk = 0.f;
#pragma unroll
    for (int w = 0; w < 8; ++w) { tq += redq[w]; tk += redk[w]; }
    const float nq = rsqrtf(tq / (float)DIM_ + 1e-6f);
    const float nk = rsqrtf(tk / (float)DIM_ + 1e-6f);

    // RoPE over 768 pairs (3 per thread)
    for (int p = tid; p < H_ * NPAIR; p += 256) {
        const int h  = p >> 6;           // head
        const int pp = p & 63;           // pair within head
        const int d0 = h * HD_ + 2 * pp;
        const float cr = freqs[s * HD_ + 2 * pp];
        const float ci = freqs[s * HD_ + 2 * pp + 1];

        float tr = qrow[d0]     * nq * gq[d0];
        float ti = qrow[d0 + 1] * nq * gq[d0 + 1];
        size_t ob = ((size_t)(b * H_ + h) * S_ + s) * HD_ + 2 * pp;
        qh[ob]     = (_Float16)(tr * cr - ti * ci);
        qh[ob + 1] = (_Float16)(tr * ci + ti * cr);

        tr = krow[d0]     * nk * gk[d0];
        ti = krow[d0 + 1] * nk * gk[d0 + 1];
        kh[ob]     = (_Float16)(tr * cr - ti * ci);
        kh[ob + 1] = (_Float16)(tr * ci + ti * cr);
    }

    // V: cast + transpose to [B,H,HD,S]
    const float* vrow = vf + (size_t)tok * DIM_;
    for (int i = tid; i < DIM_; i += 256) {
        const int h = i / HD_;
        const int d = i % HD_;
        vth[((size_t)(b * H_ + h) * HD_ + d) * S_ + s] = (_Float16)vrow[i];
    }
}

// ---------------------------------------------------------------------------
// 3) Flash attention.  grid = (S/128, B*H); 8 waves, wave owns 16 q-rows.
//    Key blocks of 32, double-buffered via async global->LDS copies.
//    Online softmax in C-layout; P transposed via wave-private LDS.
//    Output: oh [B,S,DIM] f16.
// ---------------------------------------------------------------------------
__global__ __launch_bounds__(256)
void attn_kernel(const _Float16* __restrict__ qh, const _Float16* __restrict__ kh,
                 const _Float16* __restrict__ vth, _Float16* __restrict__ oh)
{
    __shared__ __align__(16) _Float16 Ks[2][32][HD_ + 8];   // row 272 B
    __shared__ __align__(16) _Float16 Vts[2][HD_][32 + 8];  // row 80 B
    __shared__ __align__(16) _Float16 Pw[8][16][32 + 8];    // per-wave P tile

    const int tid  = threadIdx.x;
    const int lane = tid & 31;
    const int wave = tid >> 5;
    const int bh = blockIdx.y;           // 0..23
    const int b = bh / H_, h = bh % H_;

    const _Float16* qbase = qh  + (size_t)bh * S_ * HD_;
    const _Float16* kbase = kh  + (size_t)bh * S_ * HD_;
    const _Float16* vbase = vth + (size_t)bh * HD_ * S_;

    const int q0   = blockIdx.x * 128 + wave * 16;
    const int ln16 = lane & 15;
    const int half = lane >> 4;          // 0/1

    // Q fragments (A layout): 4 chunks of K=32 along HD
    half16 qfrag[4];
    {
        const _Float16* qp = qbase + (size_t)(q0 + ln16) * HD_;
#pragma unroll
        for (int f = 0; f < 4; ++f) {
            const _Float16* p = qp + f * 32;
            qfrag[f] = cat16(*(const half8*)(p + half * 8),
                             *(const half8*)(p + 16 + half * 8));
        }
    }

    // async staging of one 32-key block (K rows + V^T columns)
    const int krow_ld = tid >> 3, koff_ld = (tid & 7) * 16;  // 8 thr / K row
    const int vrow_ld = tid >> 1, voff_ld = (tid & 1) * 16;  // 2 thr / Vt row
    auto issue_kv = [&](int kb, int buf) {
        const int kst = kb * 32;
        const _Float16* kg = kbase + (size_t)(kst + krow_ld) * HD_ + koff_ld;
        async_ld_b128(&Ks[buf][krow_ld][koff_ld],     kg);
        async_ld_b128(&Ks[buf][krow_ld][koff_ld + 8], kg + 8);
        const _Float16* vg = vbase + (size_t)vrow_ld * S_ + kst + voff_ld;
        async_ld_b128(&Vts[buf][vrow_ld][voff_ld],     vg);
        async_ld_b128(&Vts[buf][vrow_ld][voff_ld + 8], vg + 8);
    };

    f32x8 acc[8];
    float rmax[8], rsum[8];
#pragma unroll
    for (int nt = 0; nt < 8; ++nt)
#pragma unroll
        for (int r = 0; r < 8; ++r) acc[nt][r] = 0.0f;
#pragma unroll
    for (int r = 0; r < 8; ++r) { rmax[r] = -1e30f; rsum[r] = 0.0f; }

    const float scale = 0.08838834764831845f;   // 1/sqrt(128)

    issue_kv(0, 0);
    int cur = 0;
    for (int kb = 0; kb < S_ / 32; ++kb) {
        wait_async0();
        __syncthreads();        // copies visible + prev-iteration reads done
        if (kb + 1 < S_ / 32) issue_kv(kb + 1, cur ^ 1);

        // --- S = Q * K^T (two 16-key tiles, 4 k-steps of 32 over HD)
        f32x8 sc[2];
#pragma unroll
        for (int nk = 0; nk < 2; ++nk) {
            f32x8 c;
#pragma unroll
            for (int r = 0; r < 8; ++r) c[r] = 0.0f;
            const int key = nk * 16 + ln16;
#pragma unroll
            for (int f = 0; f < 4; ++f) {
                half16 bf = cat16(
                    *(const half8*)&Ks[cur][key][f * 32 + half * 8],
                    *(const half8*)&Ks[cur][key][f * 32 + 16 + half * 8]);
                c = wmma_f16(qfrag[f], bf, c);
            }
#pragma unroll
            for (int r = 0; r < 8; ++r) c[r] *= scale;
            sc[nk] = c;
        }

        // --- online softmax (row m = r + half*8; reduce over 16 lanes)
        float alpha[8];
#pragma unroll
        for (int r = 0; r < 8; ++r) {
            float m = fmaxf(sc[0][r], sc[1][r]);
            m = fmaxf(m, __shfl_xor(m, 1));
            m = fmaxf(m, __shfl_xor(m, 2));
            m = fmaxf(m, __shfl_xor(m, 4));
            m = fmaxf(m, __shfl_xor(m, 8));
            const float nm = fmaxf(rmax[r], m);
            alpha[r] = __expf(rmax[r] - nm);
            rmax[r] = nm;
        }
#pragma unroll
        for (int nk = 0; nk < 2; ++nk)
#pragma unroll
            for (int r = 0; r < 8; ++r)
                sc[nk][r] = __expf(sc[nk][r] - rmax[r]);
#pragma unroll
        for (int r = 0; r < 8; ++r) {
            float sum = sc[0][r] + sc[1][r];
            sum += __shfl_xor(sum, 1);
            sum += __shfl_xor(sum, 2);
            sum += __shfl_xor(sum, 4);
            sum += __shfl_xor(sum, 8);
            rsum[r] = rsum[r] * alpha[r] + sum;
        }
#pragma unroll
        for (int nt = 0; nt < 8; ++nt)
#pragma unroll
            for (int r = 0; r < 8; ++r) acc[nt][r] *= alpha[r];

        // --- P: C-layout -> A-layout via wave-private LDS (same-wave DS in order)
#pragma unroll
        for (int nk = 0; nk < 2; ++nk)
#pragma unroll
            for (int r = 0; r < 8; ++r)
                Pw[wave][r + half * 8][nk * 16 + ln16] = (_Float16)sc[nk][r];
        __builtin_amdgcn_sched_barrier(0);

        half16 pfrag = cat16(*(const half8*)&Pw[wave][ln16][half * 8],
                             *(const half8*)&Pw[wave][ln16][16 + half * 8]);

        // --- O += P * V  (8 d-tiles of 16, K=32 keys)
#pragma unroll
        for (int nt = 0; nt < 8; ++nt) {
            const int d = nt * 16 + ln16;
            half16 vfr = cat16(*(const half8*)&Vts[cur][d][half * 8],
                               *(const half8*)&Vts[cur][d][16 + half * 8]);
            acc[nt] = wmma_f16(pfrag, vfr, acc[nt]);
        }
        cur ^= 1;
        // no trailing barrier (next top barrier provides the ordering)
    }

    // --- epilogue: O /= rowsum, write f16 [B,S,DIM]
#pragma unroll
    for (int r = 0; r < 8; ++r) {
        const float inv = 1.0f / rsum[r];
        const int srow = q0 + r + half * 8;
        _Float16* orow = oh + ((size_t)b * S_ + srow) * DIM_ + h * HD_;
#pragma unroll
        for (int nt = 0; nt < 8; ++nt)
            orow[nt * 16 + ln16] = (_Float16)(acc[nt][r] * inv);
    }
}

// ---------------------------------------------------------------------------
// Host-side launch
// ---------------------------------------------------------------------------
extern "C" void kernel_launch(void* const* d_in, const int* in_sizes, int n_in,
                              void* d_out, int out_size, void* d_ws, size_t ws_size,
                              hipStream_t stream) {
    const float* x     = (const float*)d_in[0];
    const float* freqs = (const float*)d_in[1];
    const float* wq = (const float*)d_in[2];  const float* bq = (const float*)d_in[3];
    const float* wk = (const float*)d_in[4];  const float* bk = (const float*)d_in[5];
    const float* wv = (const float*)d_in[6];  const float* bv = (const float*)d_in[7];
    const float* wo = (const float*)d_in[8];  const float* bo = (const float*)d_in[9];
    const float* gq = (const float*)d_in[10]; const float* gk = (const float*)d_in[11];
    (void)in_sizes; (void)n_in; (void)out_size; (void)ws_size;

    // workspace layout (bytes)
    const size_t SZ_XH = (size_t)M_ * DIM_ * 2;      // x f16
    const size_t SZ_W  = (size_t)DIM_ * DIM_ * 2;    // weight f16
    const size_t SZ_F  = (size_t)M_ * DIM_ * 4;      // proj f32
    const size_t SZ_H  = (size_t)M_ * DIM_ * 2;      // per-head f16

    char* ws = (char*)d_ws;
    _Float16* xh  = (_Float16*)(ws);                       size_t off = SZ_XH;
    _Float16* wqh = (_Float16*)(ws + off);                 off += SZ_W;
    _Float16* wkh = (_Float16*)(ws + off);                 off += SZ_W;
    _Float16* wvh = (_Float16*)(ws + off);                 off += SZ_W;
    _Float16* woh = (_Float16*)(ws + off);                 off += SZ_W;
    float*    qf  = (float*)(ws + off);                    off += SZ_F;
    float*    kf  = (float*)(ws + off);                    off += SZ_F;
    float*    vf  = (float*)(ws + off);                    off += SZ_F;
    _Float16* qh2 = (_Float16*)(ws + off);                 off += SZ_H;
    _Float16* kh2 = (_Float16*)(ws + off);                 off += SZ_H;
    _Float16* vth = (_Float16*)(ws + off);                 off += SZ_H;
    _Float16* oh  = (_Float16*)(ws + off);                 off += SZ_H;

    const int nX = M_ * DIM_, nW = DIM_ * DIM_;
    cvt_f32_to_f16<<<(nX + 255) / 256, 256, 0, stream>>>(x,  xh,  nX);
    cvt_f32_to_f16<<<(nW + 255) / 256, 256, 0, stream>>>(wq, wqh, nW);
    cvt_f32_to_f16<<<(nW + 255) / 256, 256, 0, stream>>>(wk, wkh, nW);
    cvt_f32_to_f16<<<(nW + 255) / 256, 256, 0, stream>>>(wv, wvh, nW);
    cvt_f32_to_f16<<<(nW + 255) / 256, 256, 0, stream>>>(wo, woh, nW);

    dim3 ggrid(DIM_ / BN, M_ / BM);     // (12, 32)
    gemm_nt_f16_f32<<<ggrid, 256, 0, stream>>>(xh, wqh, bq, qf, M_, DIM_, DIM_);
    gemm_nt_f16_f32<<<ggrid, 256, 0, stream>>>(xh, wkh, bk, kf, M_, DIM_, DIM_);
    gemm_nt_f16_f32<<<ggrid, 256, 0, stream>>>(xh, wvh, bv, vf, M_, DIM_, DIM_);

    rmsrope_qkv<<<M_, 256, 0, stream>>>(qf, kf, vf, freqs, gq, gk, qh2, kh2, vth);

    attn_kernel<<<dim3(S_ / 128, B_ * H_), 256, 0, stream>>>(qh2, kh2, vth, oh);

    gemm_nt_f16_f32<<<ggrid, 256, 0, stream>>>(oh, woh, bo, (float*)d_out,
                                               M_, DIM_, DIM_);
}
